// ModulatedConv3d_76381698392146
// MI455X (gfx1250) — matrix operand order repaired and checked by reading, compiled, not run on hardware
//
#include <hip/hip_runtime.h>
#include <hip/hip_bf16.h>

typedef __attribute__((ext_vector_type(16))) __bf16 v16bf;
typedef __attribute__((ext_vector_type(8)))  __bf16 v8bf;
typedef __attribute__((ext_vector_type(8)))  float  v8f;

#define B_    4
#define CIN   64
#define COUT  64
#define K3    27
#define WD    512
#define DD    48
#define KTOT  (CIN * K3)   // 1728, K order: k = (kd*9+kh*3+kw)*64 + cin
#define CPAD  72           // padded cin stride in LDS (144 B rows: b128-aligned)

// ---------- kernel 1: style[b][cin] = w[b] . affine_w[cin] + affine_b[cin]
__global__ void mc3d_style(const float* __restrict__ w,
                           const float* __restrict__ affine_w,
                           const float* __restrict__ affine_b,
                           float* __restrict__ style) {
  int t = blockIdx.x * blockDim.x + threadIdx.x;
  if (t >= B_ * CIN) return;
  int b = t / CIN, cin = t % CIN;
  const float* wr = w + b * WD;
  const float* ar = affine_w + cin * WD;
  float s = affine_b[cin];
  for (int i = 0; i < WD; ++i) s = fmaf(wr[i], ar[i], s);
  style[t] = s;
}

// ---------- kernel 2: modulate + demodulate, emit bf16 A rows in
// k-order (tap-major, cin-innermost): wmodA[b][cout][j*64 + cin], j=kd*9+kh*3+kw
__global__ void mc3d_wmod(const float* __restrict__ weight,
                          const float* __restrict__ style,
                          __bf16* __restrict__ wmodA) {
  int b    = blockIdx.x / COUT;
  int cout = blockIdx.x % COUT;
  int cin  = threadIdx.x;                    // 64 threads
  float st = style[b * CIN + cin];
  const float* wr = weight + (size_t)(cout * CIN + cin) * K3;
  float loc[K3];
  float ss = 0.f;
#pragma unroll
  for (int j = 0; j < K3; ++j) { loc[j] = wr[j] * st; ss = fmaf(loc[j], loc[j], ss); }
  __shared__ float red[CIN];
  __shared__ float dsh;
  red[cin] = ss;
  __syncthreads();
  if (cin == 0) {
    float tot = 0.f;
    for (int i = 0; i < CIN; ++i) tot += red[i];
    dsh = rsqrtf(tot + 1e-8f);
  }
  __syncthreads();
  float dmod = dsh;
  __bf16* op = wmodA + (size_t)(b * COUT + cout) * KTOT;
#pragma unroll
  for (int j = 0; j < K3; ++j) op[j * CIN + cin] = (__bf16)(loc[j] * dmod);
}

// ---------- kernel 3: implicit-GEMM conv, WMMA bf16 -> f32
// grid = (DD/16, DD*DD, B), block = 128 (4 waves; wave wv owns M-tile wv)
// LDS halo is cin-transposed so every B fragment is 16 contiguous bf16.
// Dual accumulators: two independent WMMA chains (cin-half 0 / 1).
__global__ void __launch_bounds__(128)
mc3d_conv(const float* __restrict__ x,
          const __bf16* __restrict__ wmodA,
          const float* __restrict__ bias,
          float* __restrict__ out) {
  const int b  = blockIdx.z;
  const int d  = blockIdx.y / DD;
  const int h  = blockIdx.y % DD;
  const int w0 = blockIdx.x * 16;

  // transposed halo: [q=kd*3+kh (9)][wj (18)][cin (CPAD)]
  __shared__ __bf16 sXT[9 * 18 * CPAD];   // 23328 B

  const float* xb = x + (size_t)b * CIN * DD * DD * DD;
  // 64*9*18 = 10368 = 81*128 : uniform, global-coalesced along wj; transpose on store
  for (int idx = threadIdx.x; idx < CIN * 9 * 18; idx += 128) {
    int cin = idx / 162;
    int rem = idx - cin * 162;
    int q   = rem / 18;                 // kd*3+kh
    int wj  = rem - q * 18;             // 0..17 -> zw = w0-1+wj
    int zd  = d + q / 3 - 1;
    int zh  = h + q % 3 - 1;
    int zw  = w0 - 1 + wj;
    float v = 0.f;
    if ((unsigned)zd < (unsigned)DD && (unsigned)zh < (unsigned)DD &&
        (unsigned)zw < (unsigned)DD)
      v = xb[((size_t)cin * DD + zd) * (DD * DD) + zh * DD + zw];
    sXT[(q * 18 + wj) * CPAD + cin] = (__bf16)v;
  }
  __syncthreads();

  const int lane  = threadIdx.x & 31;
  const int wv    = threadIdx.x >> 5;   // 0..3 -> M tile
  const int mbase = wv * 16;
  const int mrow  = lane & 15;
  const int hi    = (lane >> 4) & 1;
  const int n     = lane & 15;          // N column

  const __bf16* Ap = wmodA + (size_t)(b * COUT + mbase + mrow) * KTOT;

  v8f acc0 = {};
  v8f acc1 = {};
  for (int q = 0; q < 9; ++q) {
    // per-q bases; kw and half become pure immediate offsets below
    const __bf16* Brow  = &sXT[(q * 18 + n) * CPAD];
    const __bf16* Abase = Ap + q * 192;
#pragma unroll
    for (int kw = 0; kw < 3; ++kw) {
      const __bf16* Br = Brow + kw * CPAD;        // +144 B * kw (immediate)
      const __bf16* Ab = Abase + kw * 64;         // +128 B * kw (immediate)
      // ---- half 0: cin 0..31  (chain into acc0)
      {
        const v8bf b0 = *reinterpret_cast<const v8bf*>(Br + hi * 16);
        const v8bf b1 = *reinterpret_cast<const v8bf*>(Br + hi * 16 + 8);
        const v8bf a0 = *reinterpret_cast<const v8bf*>(Ab + hi * 16);
        const v8bf a1 = *reinterpret_cast<const v8bf*>(Ab + hi * 16 + 8);
        v16bf a  = __builtin_shufflevector(a0, a1, 0,1,2,3,4,5,6,7,8,9,10,11,12,13,14,15);
        v16bf bb = __builtin_shufflevector(b0, b1, 0,1,2,3,4,5,6,7,8,9,10,11,12,13,14,15);
        acc0 = __builtin_amdgcn_wmma_f32_16x16x32_bf16(
            false, a, false, bb, (short)0, acc0, false, false);
      }
      // ---- half 1: cin 32..63 (chain into acc1)
      {
        const v8bf b0 = *reinterpret_cast<const v8bf*>(Br + 32 + hi * 16);
        const v8bf b1 = *reinterpret_cast<const v8bf*>(Br + 32 + hi * 16 + 8);
        const v8bf a0 = *reinterpret_cast<const v8bf*>(Ab + 32 + hi * 16);
        const v8bf a1 = *reinterpret_cast<const v8bf*>(Ab + 32 + hi * 16 + 8);
        v16bf a  = __builtin_shufflevector(a0, a1, 0,1,2,3,4,5,6,7,8,9,10,11,12,13,14,15);
        v16bf bb = __builtin_shufflevector(b0, b1, 0,1,2,3,4,5,6,7,8,9,10,11,12,13,14,15);
        acc1 = __builtin_amdgcn_wmma_f32_16x16x32_bf16(
            false, a, false, bb, (short)0, acc1, false, false);
      }
    }
  }

  // D layout: VGPR r -> M = r + 8*hi, N = lane&15
#pragma unroll
  for (int rI = 0; rI < 8; ++rI) {
    int cout = mbase + hi * 8 + rI;
    out[(((size_t)(b * COUT + cout) * DD + d) * DD + h) * DD + w0 + n] =
        acc0[rI] + acc1[rI] + bias[cout];
  }
}

extern "C" void kernel_launch(void* const* d_in, const int* in_sizes, int n_in,
                              void* d_out, int out_size, void* d_ws, size_t ws_size,
                              hipStream_t stream) {
  const float* x        = (const float*)d_in[0];
  const float* w        = (const float*)d_in[1];
  const float* weight   = (const float*)d_in[2];
  const float* affine_w = (const float*)d_in[3];
  const float* affine_b = (const float*)d_in[4];
  const float* bias     = (const float*)d_in[5];
  float* out = (float*)d_out;

  float*  style = (float*)d_ws;
  __bf16* wmodA = (__bf16*)((char*)d_ws + 1024);  // B*COUT*KTOT bf16 = 884736 B

  mc3d_style<<<1, 256, 0, stream>>>(w, affine_w, affine_b, style);
  mc3d_wmod<<<B_ * COUT, CIN, 0, stream>>>(weight, style, wmodA);
  dim3 grid(DD / 16, DD * DD, B_);
  mc3d_conv<<<grid, dim3(128), 0, stream>>>(x, wmodA, bias, out);
}
